// TurbulenceLSTM_49306224558569
// MI455X (gfx1250) — matrix operand order repaired and checked by reading, not compiled, not run on hardware
//
#include <hip/hip_runtime.h>

typedef __attribute__((ext_vector_type(16))) _Float16 v16h;
typedef __attribute__((ext_vector_type(8)))  _Float16 v8h;
typedef __attribute__((ext_vector_type(8)))  float    v8f;

#define B_TOT 4096
#define T_LEN 512
#define IN_D  4
#define H_D   64
#define G_D   256      // 4*H
#define BT    16       // batch rows per workgroup (one WMMA M-tile)
#define HPAD  72       // padded row stride (halves) for h buffers (144B, 16B aligned)
#define GPAD  260      // padded row stride (floats) for gate scratch

__device__ __forceinline__ v8f wmma16(v16h a, v16h b, v8f c) {
  // D = A(16x32 f16) x B(32x16 f16) + C(16x16 f32)
  return __builtin_amdgcn_wmma_f32_16x16x32_f16(false, a, false, b, (short)0, c,
                                                false, false);
}

// A-fragment (16x32 f16) from LDS h-buffer, row-major [BT][HPAD].
// lanes 0-15: row=lane, K = {k0..k0+15}
// lanes 16-31: row=lane-16, K = {k0+16..k0+31}
__device__ __forceinline__ v16h load_afrag(const _Float16* sh, int k0, int lane) {
  const int row = lane & 15;
  const int sel = (lane & 16) ? 16 : 0;
  const _Float16* p = sh + row * HPAD + k0 + sel;
  v8h lo = *(const v8h*)(p);
  v8h hi = *(const v8h*)(p + 8);
  v16h a;
#pragma unroll
  for (int i = 0; i < 8; ++i) { a[i] = lo[i]; a[8 + i] = hi[i]; }
  return a;
}

// B-fragment (32x16 f16) built once from global f32 weights W[g][k] (row-major,
// 256x64). B[k][n] = W[g0+n][k]: lane holds column n=lane&15, contiguous K run.
__device__ __forceinline__ v16h load_bfrag_global(const float* W, int gbase,
                                                  int k0, int lane) {
  const int g = gbase + (lane & 15);
  const int k = k0 + ((lane & 16) ? 16 : 0);
  const float* p = W + g * H_D + k;
  v16h b;
#pragma unroll
  for (int i = 0; i < 16; ++i) b[i] = (_Float16)p[i];
  return b;
}

// D-fragment (16x16 f32) -> LDS gates[b][g]; lanes 0-15: n=lane, m=0..7;
// lanes 16-31: n=lane-16, m=8..15.
__device__ __forceinline__ void store_dfrag(float* sg, v8f d, int gbase, int lane) {
  const int n  = gbase + (lane & 15);
  const int mb = (lane & 16) ? 8 : 0;
#pragma unroll
  for (int m = 0; m < 8; ++m) sg[(mb + m) * GPAD + n] = d[m];
}

__device__ __forceinline__ float sigm(float v) {
  return __fdividef(1.f, 1.f + __expf(-v));
}
__device__ __forceinline__ float tanhfast(float v) {
  v = fminf(fmaxf(v, -15.f), 15.f);
  float e = __expf(2.f * v);
  return __fdividef(e - 1.f, e + 1.f);
}

extern "C" __global__ __launch_bounds__(256)
void lstm2_fused(const float* __restrict__ x,
                 const float* __restrict__ w_ih0, const float* __restrict__ w_hh0,
                 const float* __restrict__ b_ih0, const float* __restrict__ b_hh0,
                 const float* __restrict__ w_ih1, const float* __restrict__ w_hh1,
                 const float* __restrict__ b_ih1, const float* __restrict__ b_hh1,
                 const float* __restrict__ fc1_w, const float* __restrict__ fc1_b,
                 const float* __restrict__ fc2_w, const float* __restrict__ fc2_b,
                 float* __restrict__ out)
{
  __shared__ __align__(16) _Float16 shA[BT * HPAD];    // layer-0 hidden (f16)
  __shared__ __align__(16) _Float16 shB[BT * HPAD];    // layer-1 hidden (f16)
  __shared__ __align__(16) float    sgates[BT * GPAD]; // 16x256 gate scratch
  __shared__ float swih0[G_D * IN_D];

  const int tid   = threadIdx.x;
  const int lane  = tid & 31;
  const int wid   = tid >> 5;
  const int bbase = blockIdx.x * BT;

  // stage layer-0 input weights into LDS; zero hidden state
  for (int i = tid; i < G_D * IN_D; i += 256) swih0[i] = w_ih0[i];
  for (int i = tid; i < BT * HPAD; i += 256) {
    shA[i] = (_Float16)0.f;
    shB[i] = (_Float16)0.f;
  }

  // time-invariant weight B-fragments held in registers for whole T loop
  const int g0 = wid * 32;              // this wave's 32 gate columns
  v16h fhh0[2][2], fih1[2][2], fhh1[2][2];
  v8f  cb0[2], cb1[2];                  // bias folded into WMMA C operand
#pragma unroll
  for (int nt = 0; nt < 2; ++nt) {
#pragma unroll
    for (int ks = 0; ks < 2; ++ks) {
      fhh0[nt][ks] = load_bfrag_global(w_hh0, g0 + nt * 16, ks * 32, lane);
      fih1[nt][ks] = load_bfrag_global(w_ih1, g0 + nt * 16, ks * 32, lane);
      fhh1[nt][ks] = load_bfrag_global(w_hh1, g0 + nt * 16, ks * 32, lane);
    }
    const int n = g0 + nt * 16 + (lane & 15);
    const float bv0 = b_ih0[n] + b_hh0[n];
    const float bv1 = b_ih1[n] + b_hh1[n];
#pragma unroll
    for (int i = 0; i < 8; ++i) { cb0[nt][i] = bv0; cb1[nt][i] = bv1; }
  }

  // activation mapping: thread -> batch row ab, hidden cols [aj, aj+3]
  const int ab = tid >> 4;
  const int aj = (tid & 15) * 4;
  float cA[4] = {0.f, 0.f, 0.f, 0.f};   // cell state lives in registers
  float cB[4] = {0.f, 0.f, 0.f, 0.f};

  const float* xrow = x + (size_t)(bbase + ab) * T_LEN * IN_D;

  __syncthreads();

#pragma unroll 1
  for (int t = 0; t < T_LEN; ++t) {
    // prefetch next timestep's x row (contiguous) to hide global latency
    __builtin_prefetch(xrow + (t + 1) * IN_D, 0, 1);

    // ---- layer 0: gates = bias0 + hA @ Whh0^T (WMMA, K=64 in two k-steps) ----
#pragma unroll
    for (int nt = 0; nt < 2; ++nt) {
      v8f acc = cb0[nt];
      acc = wmma16(load_afrag(shA, 0,  lane), fhh0[nt][0], acc);
      acc = wmma16(load_afrag(shA, 32, lane), fhh0[nt][1], acc);
      store_dfrag(sgates, acc, g0 + nt * 16, lane);
    }
    __syncthreads();

    // ---- layer 0 activations (adds x @ w_ih0^T inline; I=4) ----
    {
      const float* xp = xrow + (size_t)t * IN_D;
      const float x0 = xp[0], x1 = xp[1], x2 = xp[2], x3 = xp[3];
      const float* gr = sgates + ab * GPAD;
#pragma unroll
      for (int jj = 0; jj < 4; ++jj) {
        const int j = aj + jj;
        const float* wi = swih0 + (size_t)j * IN_D;
        const float* wf = swih0 + (size_t)(j + 64) * IN_D;
        const float* wg = swih0 + (size_t)(j + 128) * IN_D;
        const float* wo = swih0 + (size_t)(j + 192) * IN_D;
        float gi = gr[j]       + x0*wi[0] + x1*wi[1] + x2*wi[2] + x3*wi[3];
        float gf = gr[j + 64]  + x0*wf[0] + x1*wf[1] + x2*wf[2] + x3*wf[3];
        float gg = gr[j + 128] + x0*wg[0] + x1*wg[1] + x2*wg[2] + x3*wg[3];
        float go = gr[j + 192] + x0*wo[0] + x1*wo[1] + x2*wo[2] + x3*wo[3];
        float c = sigm(gf) * cA[jj] + sigm(gi) * tanhfast(gg);
        cA[jj] = c;
        shA[ab * HPAD + j] = (_Float16)(sigm(go) * tanhfast(c));
      }
    }
    __syncthreads();

    // ---- layer 1: gates = bias1 + hA_new @ Wih1^T + hB_prev @ Whh1^T ----
#pragma unroll
    for (int nt = 0; nt < 2; ++nt) {
      v8f acc = cb1[nt];
      acc = wmma16(load_afrag(shA, 0,  lane), fih1[nt][0], acc);
      acc = wmma16(load_afrag(shA, 32, lane), fih1[nt][1], acc);
      acc = wmma16(load_afrag(shB, 0,  lane), fhh1[nt][0], acc);
      acc = wmma16(load_afrag(shB, 32, lane), fhh1[nt][1], acc);
      store_dfrag(sgates, acc, g0 + nt * 16, lane);
    }
    __syncthreads();

    // ---- layer 1 activations ----
    {
      const float* gr = sgates + ab * GPAD;
#pragma unroll
      for (int jj = 0; jj < 4; ++jj) {
        const int j = aj + jj;
        float gi = gr[j];
        float gf = gr[j + 64];
        float gg = gr[j + 128];
        float go = gr[j + 192];
        float c = sigm(gf) * cB[jj] + sigm(gi) * tanhfast(gg);
        cB[jj] = c;
        shB[ab * HPAD + j] = (_Float16)(sigm(go) * tanhfast(c));
      }
    }
    __syncthreads();
  }

  // ---- head: out = relu(relu(hB @ fc1^T + b1) @ fc2^T + b2), tiny -> VALU ----
  float* sy = sgates;  // reuse as 16x32 y buffer
  for (int e = tid; e < BT * 32; e += 256) {
    const int b = e >> 5, j = e & 31;
    float acc = fc1_b[j];
    const float*    wr = fc1_w + (size_t)j * H_D;
    const _Float16* hr = shB + b * HPAD;
#pragma unroll
    for (int k = 0; k < H_D; ++k) acc += (float)hr[k] * wr[k];
    sy[b * 32 + j] = fmaxf(acc, 0.f);
  }
  __syncthreads();
  if (tid < BT * 2) {
    const int b = tid >> 1, j = tid & 1;
    float acc = fc2_b[j];
    const float* wr = fc2_w + (size_t)j * 32;
    const float* yr = sy + b * 32;
#pragma unroll
    for (int k = 0; k < 32; ++k) acc += yr[k] * wr[k];
    out[(size_t)(bbase + b) * 2 + j] = fmaxf(acc, 0.f);
  }
}

extern "C" void kernel_launch(void* const* d_in, const int* in_sizes, int n_in,
                              void* d_out, int out_size, void* d_ws, size_t ws_size,
                              hipStream_t stream) {
  (void)in_sizes; (void)n_in; (void)d_ws; (void)ws_size; (void)out_size;
  const float* x     = (const float*)d_in[0];
  const float* w_ih0 = (const float*)d_in[1];
  const float* w_hh0 = (const float*)d_in[2];
  const float* b_ih0 = (const float*)d_in[3];
  const float* b_hh0 = (const float*)d_in[4];
  const float* w_ih1 = (const float*)d_in[5];
  const float* w_hh1 = (const float*)d_in[6];
  const float* b_ih1 = (const float*)d_in[7];
  const float* b_hh1 = (const float*)d_in[8];
  const float* fc1_w = (const float*)d_in[9];
  const float* fc1_b = (const float*)d_in[10];
  const float* fc2_w = (const float*)d_in[11];
  const float* fc2_b = (const float*)d_in[12];

  dim3 grid(B_TOT / BT);   // 256 workgroups
  dim3 block(256);         // 8 wave32 waves
  lstm2_fused<<<grid, block, 0, stream>>>(x, w_ih0, w_hh0, b_ih0, b_hh0,
                                          w_ih1, w_hh1, b_ih1, b_hh1,
                                          fc1_w, fc1_b, fc2_w, fc2_b,
                                          (float*)d_out);
}
